// MHA__11948599017860
// MI455X (gfx1250) — matrix operand order repaired and checked by reading, not compiled, not run on hardware
//
#include <hip/hip_runtime.h>
#include <cstdint>

// ---------------------------------------------------------------------------
// MHA (per-head 1x1-conv attention), flash-style, f16 WMMA on gfx1250,
// with async global->LDS staging of K/V tiles (ASYNCcnt / s_wait_asynccnt).
//   H=12 heads, Dh=64, N=3072, fp32 in/out, f16 Q/K/V intermediates in d_ws.
// ---------------------------------------------------------------------------

typedef __attribute__((ext_vector_type(16))) _Float16 v16h;
typedef __attribute__((ext_vector_type(8)))  _Float16 v8h;
typedef __attribute__((ext_vector_type(8)))  float    v8f;

#define HEADS 12
#define DMODEL 768
#define NSEQ 3072
#define DH 64

// -------------------------- projection kernel ------------------------------
// grid = (NSEQ/64, HEADS), block = 256.
// Q[h][d][n], V[h][d][n]  (row-major in n)   Kt[h][n][d]  (keys as rows)
__global__ __launch_bounds__(256) void proj_qkv_kernel(
    const float* __restrict__ x,   // [N, 1, D]
    const float* __restrict__ Wq,  // [H, DH, DH]
    const float* __restrict__ Wk,
    const float* __restrict__ Wv,
    _Float16* __restrict__ Q,      // [H, DH, N]
    _Float16* __restrict__ Kt,     // [H, N, DH]
    _Float16* __restrict__ V)      // [H, DH, N]
{
    __shared__ float xs[DH][DH + 1];   // [e][n_local], padded vs bank conflicts

    const int h  = blockIdx.y;
    const int nb = blockIdx.x * 64;
    const int tid = threadIdx.x;

    // Stage X tile: X[h][e][n] = x[n*D + h*64 + e]; coalesced over e.
#pragma unroll
    for (int i = 0; i < 16; ++i) {
        int idx = tid + i * 256;          // 0..4095
        int e  = idx & 63;
        int nl = idx >> 6;
        xs[e][nl] = x[(size_t)(nb + nl) * DMODEL + h * DH + e];
    }
    __syncthreads();

    const int d   = tid & 63;
    const int nl0 = tid >> 6;             // 0..3, each thread does 16 columns

    const float* wq = Wq + (size_t)(h * DH + d) * DH;
    const float* wk = Wk + (size_t)(h * DH + d) * DH;
    const float* wv = Wv + (size_t)(h * DH + d) * DH;

    float aq[16], ak[16], av[16];
#pragma unroll
    for (int j = 0; j < 16; ++j) { aq[j] = 0.f; ak[j] = 0.f; av[j] = 0.f; }

#pragma unroll 4
    for (int e = 0; e < DH; ++e) {
        float wqe = wq[e], wke = wk[e], wve = wv[e];
#pragma unroll
        for (int j = 0; j < 16; ++j) {
            float xv = xs[e][nl0 + 4 * j];
            aq[j] = fmaf(wqe, xv, aq[j]);
            ak[j] = fmaf(wke, xv, ak[j]);
            av[j] = fmaf(wve, xv, av[j]);
        }
    }

#pragma unroll
    for (int j = 0; j < 16; ++j) {
        int n = nb + nl0 + 4 * j;
        Q[(size_t)(h * DH + d) * NSEQ + n]  = (_Float16)aq[j];
        V[(size_t)(h * DH + d) * NSEQ + n]  = (_Float16)av[j];
        Kt[((size_t)h * NSEQ + n) * DH + d] = (_Float16)ak[j];
    }
}

// ---------------------- fragment helpers (16-bit A layout) -----------------
// A 16x32 f16: lane row = lane%16; lanes<16 hold K 0..7 (v16h[0..7]) and
// K 16..23 (v16h[8..15]); lanes>=16 hold K 8..15 and 24..31.
static __device__ inline v16h load_a_frag(const _Float16* row_base, int ko) {
    v8h lo = *(const v8h*)(row_base + ko);
    v8h hi = *(const v8h*)(row_base + ko + 16);
    v16h r;
#pragma unroll
    for (int i = 0; i < 8; ++i) { r[i] = lo[i]; r[i + 8] = hi[i]; }
    return r;
}

static __device__ inline v8f wmma_f16(v16h a, v16h b, v8f c) {
    return __builtin_amdgcn_wmma_f32_16x16x32_f16(
        /*neg_a=*/false, a, /*neg_b=*/false, b,
        /*c_mod=*/(short)0, c, /*reuse_a=*/false, /*reuse_b=*/false);
}

// One 16-byte async DMA: global -> LDS, tracked by ASYNCcnt.
static __device__ inline void async_copy_b128(const _Float16* gsrc, _Float16* ldst) {
    uint32_t lds_off = (uint32_t)(uintptr_t)ldst;   // addrspace(3) byte offset
    asm volatile("global_load_async_to_lds_b128 %0, %1, off"
                 :: "v"(lds_off), "v"(gsrc)
                 : "memory");
}

static __device__ inline void wait_async_zero() {
    asm volatile("s_wait_asynccnt 0" ::: "memory");
}

// --------------------------- attention kernel ------------------------------
// grid = (NSEQ/128, HEADS), block = 256 (8 waves). Each wave: 16 queries.
// Workgroup stages each 32-key Kt/V tile into LDS ONCE via async DMA
// (double-buffered); all 8 waves consume it. Online softmax (flash attn).
#define PLDS_STRIDE 24   // halves per row; 48B => 16B-aligned rows

__global__ __launch_bounds__(256) void flash_attn_kernel(
    const _Float16* __restrict__ Q,   // [H, DH, N]
    const _Float16* __restrict__ Kt,  // [H, N, DH]
    const _Float16* __restrict__ V,   // [H, DH, N]
    float* __restrict__ out)          // [N, 1, D]
{
    __shared__ _Float16 kbuf[2][32][DH];   // [buf][k][d]   4KB each
    __shared__ _Float16 vbuf[2][DH][32];   // [buf][d][k]   4KB each
    __shared__ _Float16 plds[8][32][PLDS_STRIDE];

    const int h    = blockIdx.y;
    const int tid  = threadIdx.x;
    const int wave = tid >> 5;
    const int lane = tid & 31;
    const int q0   = blockIdx.x * 128 + wave * 16;

    const int col  = lane & 15;              // q column (C/D layout)
    const int hi16 = (lane >= 16) ? 1 : 0;
    const int ko   = hi16 ? 8 : 0;           // A-layout K sub-offset
    const int rbase = hi16 ? 8 : 0;          // C-layout row base

    const _Float16* Qh  = Q  + (size_t)h * DH * NSEQ;
    const _Float16* Kth = Kt + (size_t)h * NSEQ * DH;
    const _Float16* Vh  = V  + (size_t)h * DH * NSEQ;

    // Per-thread source/dest coordinates for the async staging (1 B128 each).
    const int skr = tid >> 3, skc = (tid & 7) * 8;   // Kt tile: 32 x 64
    const int svd = tid >> 2, svc = (tid & 3) * 8;   // V  tile: 64 x 32

    // Q B-fragments (K-dim = d): lane holds q col = q0 + lane%16, d in table order.
    v16h qf[2];
#pragma unroll
    for (int dh = 0; dh < 2; ++dh)
#pragma unroll
        for (int s = 0; s < 16; ++s)
            qf[dh][s] = Qh[(size_t)(dh * 32 + ko + (s < 8 ? s : s + 8)) * NSEQ + q0 + col];

    v8f o[4];
#pragma unroll
    for (int t = 0; t < 4; ++t)
#pragma unroll
        for (int i = 0; i < 8; ++i) o[t][i] = 0.f;

    float m = -3.0e38f;   // running max for column q = lane%16
    float l = 0.f;        // running sum

    const float scale = 0.125f;   // 1/sqrt(64)

    // Prologue: stage first tile into buffer 0.
    async_copy_b128(Kth + (size_t)skr * DH + skc,        &kbuf[0][skr][skc]);
    async_copy_b128(Vh  + (size_t)svd * NSEQ + svc,      &vbuf[0][svd][svc]);

    int p = 0;
    for (int kb = 0; kb < NSEQ; kb += 32) {
        // Own async loads landed; barrier => everyone's landed AND everyone
        // is done reading the buffer we are about to overwrite.
        wait_async_zero();
        __syncthreads();

        if (kb + 32 < NSEQ) {
            int np = p ^ 1;
            async_copy_b128(Kth + (size_t)(kb + 32 + skr) * DH + skc,
                            &kbuf[np][skr][skc]);
            async_copy_b128(Vh + (size_t)svd * NSEQ + (kb + 32) + svc,
                            &vbuf[np][svd][svc]);
        }

        // ---- S = Kt * Q  (two 16x16 k-subtiles, contraction over d=64) ----
        v8f s[2];
#pragma unroll
        for (int sub = 0; sub < 2; ++sub) {
            const _Float16* krow = &kbuf[p][sub * 16 + col][0];
            v8f c;
#pragma unroll
            for (int i = 0; i < 8; ++i) c[i] = 0.f;
#pragma unroll
            for (int dh = 0; dh < 2; ++dh) {
                v16h a = load_a_frag(krow + dh * 32, ko);
                c = wmma_f16(a, qf[dh], c);
            }
#pragma unroll
            for (int i = 0; i < 8; ++i) s[sub][i] = c[i] * scale;
        }

        // ---- online softmax over the 32 keys of this tile ----
        float tmax = -3.0e38f;
#pragma unroll
        for (int sub = 0; sub < 2; ++sub)
#pragma unroll
            for (int i = 0; i < 8; ++i) tmax = fmaxf(tmax, s[sub][i]);
        tmax = fmaxf(tmax, __shfl_xor(tmax, 16, 32));   // join lane halves

        float mnew  = fmaxf(m, tmax);
        float alpha = __expf(m - mnew);
        m = mnew;

        float lsum = 0.f;
        _Float16 ph[2][8];
#pragma unroll
        for (int sub = 0; sub < 2; ++sub)
#pragma unroll
            for (int i = 0; i < 8; ++i) {
                float pe = __expf(s[sub][i] - mnew);
                lsum += pe;
                ph[sub][i] = (_Float16)pe;
            }
        lsum += __shfl_xor(lsum, 16, 32);
        l = l * alpha + lsum;

#pragma unroll
        for (int t = 0; t < 4; ++t)
#pragma unroll
            for (int i = 0; i < 8; ++i) o[t][i] *= alpha;

        // ---- P (C layout) -> LDS [k][q] -> B fragment ----
#pragma unroll
        for (int sub = 0; sub < 2; ++sub)
#pragma unroll
            for (int r = 0; r < 8; ++r)
                plds[wave][sub * 16 + rbase + r][col] = ph[sub][r];

        // Same-wave DS ops are in-order; compiler tracks DScnt for the use.
        v16h pb;
#pragma unroll
        for (int s = 0; s < 16; ++s) pb[s] = plds[wave][ko + (s < 8 ? s : s + 8)][col];

        // ---- O += V * P  (A = V[d,k] tiles from LDS, B = P[k,q]) ----
#pragma unroll
        for (int t = 0; t < 4; ++t) {
            const _Float16* vrow = &vbuf[p][t * 16 + col][0];
            v16h a = load_a_frag(vrow, ko);
            o[t] = wmma_f16(a, pb, o[t]);
        }

        p ^= 1;
    }

    // ---- normalize and scatter: out[q*D + h*64 + d] ----
    const float linv = 1.0f / l;
    const int qg = q0 + col;
#pragma unroll
    for (int t = 0; t < 4; ++t)
#pragma unroll
        for (int r = 0; r < 8; ++r) {
            int d = t * 16 + rbase + r;
            out[(size_t)qg * DMODEL + h * DH + d] = o[t][r] * linv;
        }
}

// ------------------------------- launcher ----------------------------------
extern "C" void kernel_launch(void* const* d_in, const int* in_sizes, int n_in,
                              void* d_out, int out_size, void* d_ws, size_t ws_size,
                              hipStream_t stream) {
    const float* x  = (const float*)d_in[0];
    const float* Wq = (const float*)d_in[1];
    const float* Wk = (const float*)d_in[2];
    const float* Wv = (const float*)d_in[3];
    float* out = (float*)d_out;

    const size_t per = (size_t)HEADS * DH * NSEQ;   // 2,359,296 f16 elems
    _Float16* Qf  = (_Float16*)d_ws;
    _Float16* Ktf = Qf + per;
    _Float16* Vf  = Ktf + per;

    dim3 pgrid(NSEQ / 64, HEADS);
    proj_qkv_kernel<<<pgrid, 256, 0, stream>>>(x, Wq, Wk, Wv, Qf, Ktf, Vf);

    dim3 agrid(NSEQ / 128, HEADS);
    flash_attn_kernel<<<agrid, 256, 0, stream>>>(Qf, Ktf, Vf, out);
}
